// TransformerBlock_43018392437032
// MI455X (gfx1250) — matrix operand order repaired and checked by reading, not compiled, not run on hardware
//
#include <hip/hip_runtime.h>

// ---------------------------------------------------------------------------
// MI455X (gfx1250) transformer block, bf16 WMMA everywhere.
// ~1.2 TFLOP vs ~20us of HBM traffic at 23.3 TB/s -> strongly compute bound,
// so all five GEMMs + both attention matmuls run on v_wmma_f32_16x16x32_bf16.
// GEMM tiles are DMA'd into LDS by the Tensor Data Mover (tensor_load_to_lds,
// TENSORcnt), double-buffered so the TDM fills tile k+1 while WMMAs consume
// tile k. Flash attention stages V tiles with GLOBAL_LOAD_ASYNC_TO_LDS_B128
// (ASYNCcnt). bf16 weights (134 MB) fit in the 192 MB L2.
// ---------------------------------------------------------------------------

typedef __bf16 bf16_t;
typedef unsigned int u32;
typedef __attribute__((ext_vector_type(16))) __bf16 bf16x16;
typedef __attribute__((ext_vector_type(8)))  __bf16 bf16x8;
typedef __attribute__((ext_vector_type(8)))  float  f32x8;
typedef __attribute__((ext_vector_type(4)))  u32    u32x4;
typedef __attribute__((ext_vector_type(8)))  u32    u32x8;

#define D_MODEL 2048
#define N_HEADS 16
#define D_HEAD  128
#define D_FF    8192
#define SEQ     2048
#define BATCH   4
#define NROWS   (BATCH * SEQ)   // 8192 token rows

// ---------------------------------------------------------------------------
// CDNA5 data movers.
// - async copy: global -> LDS, 16B per lane, ASYNCcnt / s_wait_asynccnt.
//   VDST VGPR = LDS byte address (low 32 bits of a generic pointer to
//   __shared__ are the hardware LDS address, ISA 10.2).
// - TDM: tensor_load_to_lds with a D# built in SGPRs (CDNA5 ISA):
//   group0 = {count=1 | lds_addr | global_addr[56:0] | type=2}
//   group1 = {mask=0, data_size=3 (8B), tensor/tile dims, dim0 stride}
//   groups 2/3 zero (2D tile). Tracked by TENSORcnt.
// ---------------------------------------------------------------------------
__device__ __forceinline__ unsigned lds_addr(const void* p) {
  return (unsigned)(size_t)p;
}
__device__ __forceinline__ void async_ld_b128(unsigned ldsOff, const void* g) {
  asm volatile("global_load_async_to_lds_b128 %0, %1, off"
               : : "v"(ldsOff), "v"(g) : "memory");
}
__device__ __forceinline__ void wait_async0() {
  asm volatile("s_wait_asynccnt 0" ::: "memory");
}

// 2D tile DMA, all dims in 8-byte elements (data_size = 3).
__device__ __forceinline__ void tdm_load_2d(unsigned ldsOff, const void* gaddr,
                                            u32 rows, u32 row_units,
                                            u32 stride_units) {
  unsigned long long ga = (unsigned long long)gaddr;
  u32x4 g0;
  g0[0] = 1u;                                            // count=1, user D#
  g0[1] = ldsOff;                                        // lds_addr
  g0[2] = (u32)ga;                                       // global_addr[31:0]
  g0[3] = (u32)((ga >> 32) & 0x01FFFFFFu) | (2u << 30);  // addr[56:32]|type=2
  u32x8 g1;
  g1[0] = 3u << 16;                                      // data_size = 8B
  g1[1] = (row_units & 0xFFFFu) << 16;                   // tensor_dim0 lo
  g1[2] = (row_units >> 16) | ((rows & 0xFFFFu) << 16);  // td0 hi | td1 lo
  g1[3] = (rows >> 16) | (row_units << 16);              // td1 hi | tile_dim0
  g1[4] = rows;                                          // tile_dim1 (td2=0)
  g1[5] = stride_units;                                  // dim0 stride lo
  g1[6] = 0;                                             // stride hi | d1 stride
  g1[7] = 0;
  u32x4 gz = {0u, 0u, 0u, 0u};                           // 2D: groups 2/3 zero
  asm volatile("tensor_load_to_lds %0, %1, %2, %3"
               : : "s"(g0), "s"(g1), "s"(gz), "s"(gz) : "memory");
}
__device__ __forceinline__ void wait_tensor0() {
  __builtin_amdgcn_s_wait_tensorcnt(0);
}

// ---------------------------------------------------------------------------
// WMMA fragment helpers (A/B in this file's contiguous k order; C/D per CDNA5 ISA)
// A 16x32 bf16: lane L row M=L%16; K=(L/16)*16..+15 (same k order as B)
// -> two contiguous 8-element loads.
// B 32x16 bf16: lane L col N=L%16; K=(L/16)*16..+15 -> one contiguous
// 16-element load from a row-major [N,K] matrix.
// C/D 16x16 f32: lane L col N=L%16, VGPR i row M = i + 8*(L/16).
// ---------------------------------------------------------------------------
__device__ __forceinline__ bf16x16 frag_a_from(const bf16_t* p) {
  bf16x8 lo = *(const bf16x8*)(p);
  bf16x8 hi = *(const bf16x8*)(p + 8);
  bf16x16 r;
#pragma unroll
  for (int i = 0; i < 8; ++i) { r[i] = lo[i]; r[i + 8] = hi[i]; }
  return r;
}
__device__ __forceinline__ bf16x16 load_a_frag(const bf16_t* A, int lda,
                                               int row0, int k0, int lane) {
  int m  = row0 + (lane & 15);
  int kb = k0 + (lane >> 4) * 16;
  return frag_a_from(A + (size_t)m * lda + kb);
}
__device__ __forceinline__ bf16x16 load_b_frag(const bf16_t* W, int ldw,
                                               int col0, int k0, int lane) {
  int n  = col0 + (lane & 15);
  int kh = k0 + (lane >> 4) * 16;
  return *(const bf16x16*)(W + (size_t)n * ldw + kh);
}
__device__ __forceinline__ f32x8 wmma_bf16(bf16x16 a, bf16x16 b, f32x8 c) {
  return __builtin_amdgcn_wmma_f32_16x16x32_bf16(false, a, false, b,
                                                 (short)0, c, false, false);
}

__device__ __forceinline__ float red16max(float v) {
#pragma unroll
  for (int off = 1; off < 16; off <<= 1) v = fmaxf(v, __shfl_xor(v, off, 32));
  return v;  // xor masks 1/2/4/8 stay inside each 16-lane half (one C row)
}
__device__ __forceinline__ float red16sum(float v) {
#pragma unroll
  for (int off = 1; off < 16; off <<= 1) v += __shfl_xor(v, off, 32);
  return v;
}

// ---------------------------------------------------------------------------
// fp32 -> bf16 cast (weights; once per launch, deterministic)
// ---------------------------------------------------------------------------
__global__ void cast_f32_to_bf16(const float* __restrict__ s,
                                 bf16_t* __restrict__ d, long n) {
  long i = ((long)blockIdx.x * blockDim.x + threadIdx.x) * 4;
  if (i + 3 < n) {
    float4 v = *(const float4*)(s + i);
    d[i + 0] = (bf16_t)v.x; d[i + 1] = (bf16_t)v.y;
    d[i + 2] = (bf16_t)v.z; d[i + 3] = (bf16_t)v.w;
  }
}

// ---------------------------------------------------------------------------
// RMSNorm: one 256-thread block per token row (2048 cols), LDS reduction.
// ---------------------------------------------------------------------------
__global__ __launch_bounds__(256)
void rmsnorm_to_bf16(const float* __restrict__ x, const float* __restrict__ w,
                     bf16_t* __restrict__ out) {
  __shared__ float red[256];
  int row = blockIdx.x, t = threadIdx.x;
  const float* xr = x + (size_t)row * D_MODEL;
  float4 a = ((const float4*)xr)[2 * t];
  float4 b = ((const float4*)xr)[2 * t + 1];
  float ss = a.x * a.x + a.y * a.y + a.z * a.z + a.w * a.w +
             b.x * b.x + b.y * b.y + b.z * b.z + b.w * b.w;
  red[t] = ss;
  __syncthreads();
#pragma unroll
  for (int s2 = 128; s2 > 0; s2 >>= 1) {
    if (t < s2) red[t] += red[t + s2];
    __syncthreads();
  }
  float rms = rsqrtf(red[0] * (1.0f / D_MODEL) + 1e-5f);
  int c = t * 8;
  const float* wp = w + c;
  bf16_t* op = out + (size_t)row * D_MODEL + c;
  op[0] = (bf16_t)(a.x * rms * wp[0]); op[1] = (bf16_t)(a.y * rms * wp[1]);
  op[2] = (bf16_t)(a.z * rms * wp[2]); op[3] = (bf16_t)(a.w * rms * wp[3]);
  op[4] = (bf16_t)(b.x * rms * wp[4]); op[5] = (bf16_t)(b.y * rms * wp[5]);
  op[6] = (bf16_t)(b.z * rms * wp[6]); op[7] = (bf16_t)(b.w * rms * wp[7]);
}

// ---------------------------------------------------------------------------
// TDM-staged WMMA GEMM: C[M,N] = A[M,K](bf16) * W[N,K](bf16)^T
// Block = 256 threads / 8 waves, block tile 64(M) x 128(N), K-step 32.
// Wave 0 issues two tensor_load_to_lds per K-step (A 64x32 = 4 KB,
// B 128x32 = 8 KB), double buffered; every wave owns a 32x32 accumulator
// (2 A-frags x 2 B-frags = 4 WMMAs per step). ~42 FLOP per global byte,
// and the DMA engine leaves the VALU/VMEM pipes free for WMMA.
// EPI 0: store bf16. EPI 1: fp32 residual add (OUT may alias RES).
// ---------------------------------------------------------------------------
#define BLK_M 64
#define BLK_N 128
#define BLK_K 32

template <int EPI>
__global__ __launch_bounds__(256)
void gemm_bf16_wmma_tdm(const bf16_t* __restrict__ A,
                        const bf16_t* __restrict__ W,
                        void* __restrict__ OUTv,
                        const float* __restrict__ RES,
                        int M, int N, int K) {
  __shared__ alignas(16) bf16_t sA[2][BLK_M * BLK_K];   // 2 x 4 KB
  __shared__ alignas(16) bf16_t sB[2][BLK_N * BLK_K];   // 2 x 8 KB

  int t = threadIdx.x;
  int lane = t & 31, wid = t >> 5;
  int waveM = wid & 1, waveN = wid >> 1;        // 2 x 4 wave grid
  int nn = lane & 15, half = lane >> 4;
  int rowBase = blockIdx.y * BLK_M;
  int colBase = blockIdx.x * BLK_N;

  const bf16_t* aTile = A + (size_t)rowBase * K;   // + k0 per step
  const bf16_t* bTile = W + (size_t)colBase * K;
  unsigned ldsA[2] = { lds_addr(&sA[0][0]), lds_addr(&sA[1][0]) };
  unsigned ldsB[2] = { lds_addr(&sB[0][0]), lds_addr(&sB[1][0]) };
  const u32 strideU = (u32)(K / 4);    // row stride, 8-byte units
  const u32 rowU = BLK_K / 4;          // 32 bf16 = 8 units per tile row

  f32x8 acc[2][2];
#pragma unroll
  for (int i = 0; i < 2; ++i)
#pragma unroll
    for (int j = 0; j < 2; ++j)
      acc[i][j] = (f32x8){0.f, 0.f, 0.f, 0.f, 0.f, 0.f, 0.f, 0.f};

  // prologue: DMA buffer 0
  if (wid == 0) {
    tdm_load_2d(ldsA[0], aTile, BLK_M, rowU, strideU);
    tdm_load_2d(ldsB[0], bTile, BLK_N, rowU, strideU);
    wait_tensor0();
  }
  __syncthreads();

  int nk = K / BLK_K;
  for (int ki = 0; ki < nk; ++ki) {
    int buf = ki & 1;
    if (wid == 0 && ki + 1 < nk) {   // DMA the other buffer while computing
      int k0 = (ki + 1) * BLK_K;
      tdm_load_2d(ldsA[buf ^ 1], aTile + k0, BLK_M, rowU, strideU);
      tdm_load_2d(ldsB[buf ^ 1], bTile + k0, BLK_N, rowU, strideU);
    }
    const bf16_t* As = sA[buf];
    const bf16_t* Bs = sB[buf];
    bf16x16 af[2], bfr[2];
#pragma unroll
    for (int i = 0; i < 2; ++i) {
      int r = waveM * 32 + i * 16 + nn;
      af[i] = frag_a_from(As + r * BLK_K + half * 16);
      int c = waveN * 32 + i * 16 + nn;
      bfr[i] = *(const bf16x16*)(Bs + c * BLK_K + half * 16);
    }
    acc[0][0] = wmma_bf16(af[0], bfr[0], acc[0][0]);
    acc[0][1] = wmma_bf16(af[0], bfr[1], acc[0][1]);
    acc[1][0] = wmma_bf16(af[1], bfr[0], acc[1][0]);
    acc[1][1] = wmma_bf16(af[1], bfr[1], acc[1][1]);
    if (wid == 0) wait_tensor0();
    __syncthreads();
  }

#pragma unroll
  for (int i = 0; i < 2; ++i)
#pragma unroll
    for (int j = 0; j < 2; ++j)
#pragma unroll
      for (int e = 0; e < 8; ++e) {
        int row = rowBase + waveM * 32 + i * 16 + e + half * 8;
        int col = colBase + waveN * 32 + j * 16 + nn;
        size_t off = (size_t)row * N + col;
        if (EPI == 0) {
          ((bf16_t*)OUTv)[off] = (bf16_t)acc[i][j][e];
        } else {
          ((float*)OUTv)[off] = RES[off] + acc[i][j][e];
        }
      }
}

// ---------------------------------------------------------------------------
// Fused SwiGLU dual GEMM: out = silu(A*W1^T) * (A*W3^T), bf16 out.
// Direct-from-L2 streaming (bf16 W1/W3 fit in the 192 MB L2); shared A
// fragment feeds two B streams.
// ---------------------------------------------------------------------------
__global__ __launch_bounds__(256)
void gemm_bf16_wmma_swiglu(const bf16_t* __restrict__ A,
                           const bf16_t* __restrict__ W1,
                           const bf16_t* __restrict__ W3,
                           bf16_t* __restrict__ OUT, int M, int N, int K) {
  int lane = threadIdx.x & 31, wid = threadIdx.x >> 5;
  int row0 = (blockIdx.y * 2 + (wid & 1)) * 16;
  int col0 = (blockIdx.x * 4 + (wid >> 1)) * 16;
  int nn = lane & 15, half = lane >> 4;

  const bf16_t* arow  = A  + (size_t)(row0 + nn) * K;
  const bf16_t* w1row = W1 + (size_t)(col0 + nn) * K;
  const bf16_t* w3row = W3 + (size_t)(col0 + nn) * K;
  f32x8 accg = {0.f, 0.f, 0.f, 0.f, 0.f, 0.f, 0.f, 0.f};
  f32x8 accu = {0.f, 0.f, 0.f, 0.f, 0.f, 0.f, 0.f, 0.f};

  for (int k0 = 0; k0 < K; k0 += 32) {
    __builtin_prefetch(arow  + k0 + 256, 0, 3);
    __builtin_prefetch(w1row + k0 + 256, 0, 3);
    __builtin_prefetch(w3row + k0 + 256, 0, 3);
    bf16x16 a  = load_a_frag(A, K, row0, k0, lane);
    bf16x16 b1 = load_b_frag(W1, K, col0, k0, lane);
    accg = wmma_bf16(a, b1, accg);
    bf16x16 b3 = load_b_frag(W3, K, col0, k0, lane);
    accu = wmma_bf16(a, b3, accu);
  }

#pragma unroll
  for (int i = 0; i < 8; ++i) {
    int row = row0 + i + half * 8;
    int col = col0 + nn;
    float g = accg[i], u = accu[i];
    float sig = 1.0f / (1.0f + __expf(-g));
    OUT[(size_t)row * N + col] = (bf16_t)(g * sig * u);
  }
}

// ---------------------------------------------------------------------------
// RoPE in place on bf16 q or k ([B,S,H*Dh] layout, pairs (2i,2i+1)).
// ---------------------------------------------------------------------------
__global__ __launch_bounds__(256)
void rope_inplace(bf16_t* __restrict__ qk, const int* __restrict__ pos) {
  int idx = blockIdx.x * blockDim.x + threadIdx.x;
  int i = idx & 63;
  int h = (idx >> 6) & (N_HEADS - 1);
  int s = (idx >> 10) & (SEQ - 1);
  int b = idx >> 21;
  float inv = __powf(10000.0f, -(float)(2 * i) * (1.0f / D_HEAD));
  float ang = (float)pos[s] * inv;
  float sn, cs;
  __sincosf(ang, &sn, &cs);
  size_t off = ((size_t)(b * SEQ + s)) * D_MODEL + h * D_HEAD + 2 * i;
  float x1 = (float)qk[off], x2 = (float)qk[off + 1];
  qk[off]     = (bf16_t)(x1 * cs - x2 * sn);
  qk[off + 1] = (bf16_t)(x1 * sn + x2 * cs);
}

// ---------------------------------------------------------------------------
// Flash attention: one wave per (batch, head, 16-row query tile).
// Q fragments in registers; 32 keys/iter (8 QK^T WMMAs); online softmax with
// half-wave shuffles; P restaged via LDS (C->A layout); V tile async-copied
// into LDS; P.V on 8 WMMA accumulators covering the 128-wide head.
// ---------------------------------------------------------------------------
__global__ __launch_bounds__(32)
void flash_attn(const bf16_t* __restrict__ Q, const bf16_t* __restrict__ Km,
                const bf16_t* __restrict__ V, bf16_t* __restrict__ O) {
  __shared__ alignas(16) bf16_t p_lds[16 * 32];     // probs tile
  __shared__ alignas(16) bf16_t v_lds[32 * 128];    // V tile (32 keys x 128)

  int lane = threadIdx.x;
  int nn = lane & 15, half = lane >> 4;
  int qrow0 = blockIdx.x * 16;
  int h = blockIdx.y, b = blockIdx.z;
  size_t base = ((size_t)b * SEQ) * D_MODEL + (size_t)h * D_HEAD;
  const bf16_t* Qb = Q + base;
  const bf16_t* Kb = Km + base;
  const bf16_t* Vb = V + base;

  bf16x16 qa[4];
#pragma unroll
  for (int t = 0; t < 4; ++t) qa[t] = load_a_frag(Qb, D_MODEL, qrow0, t * 32, lane);

  f32x8 o[8];
  float rmax[8], rsum[8];
#pragma unroll
  for (int i = 0; i < 8; ++i) {
    o[i] = (f32x8){0.f, 0.f, 0.f, 0.f, 0.f, 0.f, 0.f, 0.f};
    rmax[i] = -1e30f;
    rsum[i] = 0.f;
  }
  const float scale = 0.08838834764831845f;  // 1/sqrt(128)
  int jend = qrow0 + 16;                     // causal: keys < jend

  unsigned vL = lds_addr(&v_lds[lane * 128]);

  for (int j0 = 0; j0 < jend; j0 += 32) {
    // async-stage 32 V rows into LDS (one row per lane, 128 bf16 = 16 ops)
    {
      const bf16_t* vr = Vb + (size_t)(j0 + lane) * D_MODEL;
#pragma unroll
      for (int c = 0; c < 128; c += 8) async_ld_b128(vL + c * 2, vr + c);
    }

    // scores for keys j0..j0+31 (two 16-wide tiles) while V streams in
    f32x8 s0 = {0.f, 0.f, 0.f, 0.f, 0.f, 0.f, 0.f, 0.f};
    f32x8 s1 = {0.f, 0.f, 0.f, 0.f, 0.f, 0.f, 0.f, 0.f};
#pragma unroll
    for (int t = 0; t < 4; ++t) {
      bf16x16 kb0 = load_b_frag(Kb, D_MODEL, j0, t * 32, lane);
      s0 = wmma_bf16(qa[t], kb0, s0);
      bf16x16 kb1 = load_b_frag(Kb, D_MODEL, j0 + 16, t * 32, lane);
      s1 = wmma_bf16(qa[t], kb1, s1);
    }

    // online softmax update, per C-tile row (row = qrow0 + i + 8*half)
    float corr[8];
#pragma unroll
    for (int i = 0; i < 8; ++i) {
      int row = qrow0 + i + half * 8;
      int c0 = j0 + nn, c1 = j0 + 16 + nn;
      float v0 = s0[i] * scale; if (c0 > row) v0 = -1e30f;
      float v1 = s1[i] * scale; if (c1 > row) v1 = -1e30f;
      float mx = red16max(fmaxf(v0, v1));
      float nm = fmaxf(rmax[i], mx);
      float cr = __expf(rmax[i] - nm);
      rmax[i] = nm;
      float p0 = __expf(v0 - nm);
      float p1 = __expf(v1 - nm);
      float ps = red16sum(p0 + p1);
      rsum[i] = rsum[i] * cr + ps;
      corr[i] = cr;
      p_lds[(i + half * 8) * 32 + nn]      = (bf16_t)p0;
      p_lds[(i + half * 8) * 32 + 16 + nn] = (bf16_t)p1;
    }
#pragma unroll
    for (int nt = 0; nt < 8; ++nt)
#pragma unroll
      for (int i = 0; i < 8; ++i) o[nt][i] *= corr[i];

    wait_async0();      // V tile resident
    __syncthreads();    // p_lds + v_lds visible wave-wide

    // reload P as an A fragment (16x32 bf16, A layout)
    bf16x16 pa;
    {
      int kb = half * 16;
#pragma unroll
      for (int t = 0; t < 8; ++t) {
        pa[t]     = p_lds[nn * 32 + kb + t];
        pa[t + 8] = p_lds[nn * 32 + kb + 8 + t];
      }
    }
    // P(16x32) . V(32x128): 8 output tiles across the head dim
#pragma unroll
    for (int nt = 0; nt < 8; ++nt) {
      bf16x16 vb;
      int d0 = nt * 16 + nn;
      int kh = half * 16;
#pragma unroll
      for (int t = 0; t < 16; ++t) vb[t] = v_lds[(kh + t) * 128 + d0];
      o[nt] = wmma_bf16(pa, vb, o[nt]);
    }
    __syncthreads();
  }

  // normalize and store (bf16, [B,S,D] with head-contiguous dims)
#pragma unroll
  for (int i = 0; i < 8; ++i) {
    int row = qrow0 + i + half * 8;
    float inv = 1.0f / rsum[i];
    bf16_t* op = O + ((size_t)(b * SEQ + row)) * D_MODEL + (size_t)h * D_HEAD + nn;
#pragma unroll
    for (int nt = 0; nt < 8; ++nt) op[nt * 16] = (bf16_t)(o[nt][i] * inv);
  }
}

// ---------------------------------------------------------------------------
// Host orchestration
// ---------------------------------------------------------------------------
extern "C" void kernel_launch(void* const* d_in, const int* in_sizes, int n_in,
                              void* d_out, int out_size, void* d_ws, size_t ws_size,
                              hipStream_t stream) {
  const float* x   = (const float*)d_in[0];
  const int*   pos = (const int*)d_in[1];
  const float* ln1 = (const float*)d_in[2];
  const float* ln2 = (const float*)d_in[3];
  const float* wq  = (const float*)d_in[4];
  const float* wk  = (const float*)d_in[5];
  const float* wv  = (const float*)d_in[6];
  const float* wo  = (const float*)d_in[7];
  const float* w1  = (const float*)d_in[8];
  const float* w2  = (const float*)d_in[9];
  const float* w3  = (const float*)d_in[10];
  float* out = (float*)d_out;

  char* ws = (char*)d_ws;
  size_t off = 0;
  auto alloc = [&](size_t nelem) -> bf16_t* {
    bf16_t* p = (bf16_t*)(ws + off);
    off += nelem * sizeof(bf16_t);
    return p;
  };
  bf16_t* wq_b = alloc((size_t)D_MODEL * D_MODEL);
  bf16_t* wk_b = alloc((size_t)D_MODEL * D_MODEL);
  bf16_t* wv_b = alloc((size_t)D_MODEL * D_MODEL);
  bf16_t* wo_b = alloc((size_t)D_MODEL * D_MODEL);
  bf16_t* w1_b = alloc((size_t)D_FF * D_MODEL);
  bf16_t* w3_b = alloc((size_t)D_FF * D_MODEL);
  bf16_t* w2_b = alloc((size_t)D_MODEL * D_FF);
  bf16_t* h_b  = alloc((size_t)NROWS * D_MODEL);
  bf16_t* q_b  = alloc((size_t)NROWS * D_MODEL);
  bf16_t* k_b  = alloc((size_t)NROWS * D_MODEL);
  bf16_t* v_b  = alloc((size_t)NROWS * D_MODEL);
  bf16_t* ao_b = alloc((size_t)NROWS * D_MODEL);
  bf16_t* gu_b = alloc((size_t)NROWS * D_FF);
  (void)ws_size; (void)in_sizes; (void)n_in; (void)out_size;

  auto cast = [&](const float* s, bf16_t* d, long n) {
    cast_f32_to_bf16<<<dim3((unsigned)((n / 4 + 255) / 256)), dim3(256), 0, stream>>>(s, d, n);
  };
  cast(wq, wq_b, (long)D_MODEL * D_MODEL);
  cast(wk, wk_b, (long)D_MODEL * D_MODEL);
  cast(wv, wv_b, (long)D_MODEL * D_MODEL);
  cast(wo, wo_b, (long)D_MODEL * D_MODEL);
  cast(w1, w1_b, (long)D_FF * D_MODEL);
  cast(w3, w3_b, (long)D_FF * D_MODEL);
  cast(w2, w2_b, (long)D_MODEL * D_FF);

  // h = rmsnorm(x) (bf16)
  rmsnorm_to_bf16<<<dim3(NROWS), dim3(256), 0, stream>>>(x, ln1, h_b);

  auto agrid = [](int M, int N) { return dim3((unsigned)(N / BLK_N), (unsigned)(M / BLK_M)); };
  dim3 gthr(256);

  // q, k, v projections (bf16 out)
  gemm_bf16_wmma_tdm<0><<<agrid(NROWS, D_MODEL), gthr, 0, stream>>>(h_b, wq_b, q_b, nullptr, NROWS, D_MODEL, D_MODEL);
  gemm_bf16_wmma_tdm<0><<<agrid(NROWS, D_MODEL), gthr, 0, stream>>>(h_b, wk_b, k_b, nullptr, NROWS, D_MODEL, D_MODEL);
  gemm_bf16_wmma_tdm<0><<<agrid(NROWS, D_MODEL), gthr, 0, stream>>>(h_b, wv_b, v_b, nullptr, NROWS, D_MODEL, D_MODEL);

  // RoPE on q and k
  int rope_total = BATCH * SEQ * N_HEADS * (D_HEAD / 2);
  rope_inplace<<<dim3((unsigned)(rope_total / 256)), dim3(256), 0, stream>>>(q_b, pos);
  rope_inplace<<<dim3((unsigned)(rope_total / 256)), dim3(256), 0, stream>>>(k_b, pos);

  // flash attention -> ao (bf16)
  flash_attn<<<dim3(SEQ / 16, N_HEADS, BATCH), dim3(32), 0, stream>>>(q_b, k_b, v_b, ao_b);

  // x1 = x + ao . Wo^T   (fp32, into d_out)
  gemm_bf16_wmma_tdm<1><<<agrid(NROWS, D_MODEL), gthr, 0, stream>>>(ao_b, wo_b, out, x, NROWS, D_MODEL, D_MODEL);

  // h2 = rmsnorm(x1)
  rmsnorm_to_bf16<<<dim3(NROWS), dim3(256), 0, stream>>>(out, ln2, h_b);

  // gu = silu(h2.W1^T) * (h2.W3^T)  (bf16)
  gemm_bf16_wmma_swiglu<<<dim3((unsigned)(D_FF / 64), (unsigned)(NROWS / 32)), gthr, 0, stream>>>(h_b, w1_b, w3_b, gu_b, NROWS, D_FF, D_MODEL);

  // out = x1 + gu . W2^T  (fp32; OUT aliases RES = d_out, read-then-write)
  gemm_bf16_wmma_tdm<1><<<agrid(NROWS, D_MODEL), gthr, 0, stream>>>(gu_b, w2_b, out, out, NROWS, D_MODEL, D_FF);
}
